// MultiheadAttention_71270687310558
// MI455X (gfx1250) — matrix operand order repaired and checked by reading, not compiled, not run on hardware
//
#include <hip/hip_runtime.h>
#include <hip/hip_bf16.h>

// ---- problem constants (from reference) ----
#define BB 2
#define SS 2048
#define EE 1024
#define HH 16
#define HD 64
#define MM (BB * SS)      // 4096

typedef __attribute__((ext_vector_type(16))) __bf16 v16bf;
typedef __attribute__((ext_vector_type(8)))  __bf16 v8bf;
typedef __attribute__((ext_vector_type(8)))  float  v8f;

// fp32 -> bf16, round-half-up (1 add + high-half move)
__device__ __forceinline__ __bf16 f2bf(float x) {
    unsigned u = __builtin_bit_cast(unsigned, x) + 0x8000u;
    return __builtin_bit_cast(__bf16, (unsigned short)(u >> 16));
}

// Pack two fp32 into two bf16 in one dword: 2 adds + 1 v_perm_b32.
// v_perm: src1 supplies byte ids 0-3, src0 supplies 4-7.
// result[15:0] = hi16(ux), result[31:16] = hi16(uy)
__device__ __forceinline__ unsigned pack2bf(float x, float y) {
    unsigned ux = __builtin_bit_cast(unsigned, x) + 0x8000u;
    unsigned uy = __builtin_bit_cast(unsigned, y) + 0x8000u;
    return __builtin_amdgcn_perm(uy, ux, 0x07060302u);
}

// 8 fp32 (two float4) -> 4 packed dwords (8 bf16)
__device__ __forceinline__ uint4 cvt8_pk(float4 a, float4 b) {
    uint4 r;
    r.x = pack2bf(a.x, a.y);
    r.y = pack2bf(a.z, a.w);
    r.z = pack2bf(b.x, b.y);
    r.w = pack2bf(b.z, b.w);
    return r;
}

__device__ __forceinline__ v16bf cat16(v8bf lo, v8bf hi) {
    return __builtin_shufflevector(lo, hi, 0, 1, 2, 3, 4, 5, 6, 7,
                                   8, 9, 10, 11, 12, 13, 14, 15);
}

// A-matrix fragment (16x32 bf16): i<8 -> k=half*8+i, i>=8 -> k=16+half*8+(i-8)
__device__ __forceinline__ v16bf lda_frag(const __bf16* p, int half) {
    v8bf lo = *(const v8bf*)(p + half * 8);
    v8bf hi = *(const v8bf*)(p + 16 + half * 8);
    return cat16(lo, hi);
}

// B-matrix fragment (32x16 bf16): i<8 -> k=half*8+i, i>=8 -> k=16+half*8+(i-8) (tile stored [n][k])
__device__ __forceinline__ v16bf ldb_frag(const __bf16* p, int half) {
    v8bf lo = *(const v8bf*)(p + half * 8);
    v8bf hi = *(const v8bf*)(p + 16 + half * 8);
    return cat16(lo, hi);
}

#define WMMA_BF16(A, B, C) \
    __builtin_amdgcn_wmma_f32_16x16x32_bf16(false, (A), false, (B), (short)0, (C), false, false)

// =====================================================================
// Kernel 1: qkv = query @ qkv_w^T + qkv_b -> Q/K/V [B,H,S,HD] bf16
// (Q pre-scaled by 1/sqrt(HD)). Block: 128 thr, 128Mx64N tile,
// 8 WMMA per 32-k step.
// =====================================================================
__global__ void qkv_gemm_wmma(const float* __restrict__ X,
                              const float* __restrict__ W,
                              const float* __restrict__ bias,
                              __bf16* __restrict__ Qb,
                              __bf16* __restrict__ Kb,
                              __bf16* __restrict__ Vb) {
    __shared__ __attribute__((aligned(16))) __bf16 As[128][48];  // [m][k]
    __shared__ __attribute__((aligned(16))) __bf16 Bs[64][48];   // [n][k]

    const int tid  = threadIdx.x;
    const int lane = tid & 31;
    const int w    = tid >> 5;
    const int half = lane >> 4;
    const int l16  = lane & 15;
    const int n0   = blockIdx.x * 64;
    const int m0   = blockIdx.y * 128;

    v8f acc[2][4] = {};

    for (int k0 = 0; k0 < EE; k0 += 32) {
#pragma unroll
        for (int u = tid; u < 256; u += 128) {   // A: 128 rows x 2 chunks
            int m = u >> 1, c = u & 1;
            const float4* s4 = (const float4*)(X + (size_t)(m0 + m) * EE + k0 + c * 16);
            float4 x0 = s4[0], x1 = s4[1], x2 = s4[2], x3 = s4[3];
            *(uint4*)&As[m][c * 16]     = cvt8_pk(x0, x1);
            *(uint4*)&As[m][c * 16 + 8] = cvt8_pk(x2, x3);
        }
        {   // B [n][k]: straight copy of W rows
            int n = tid >> 1, c = tid & 1;
            const float4* s4 = (const float4*)(W + (size_t)(n0 + n) * EE + k0 + c * 16);
            float4 x0 = s4[0], x1 = s4[1], x2 = s4[2], x3 = s4[3];
            *(uint4*)&Bs[n][c * 16]     = cvt8_pk(x0, x1);
            *(uint4*)&Bs[n][c * 16 + 8] = cvt8_pk(x2, x3);
        }
        if (k0 + 32 < EE) {   // prefetch next k-tile
            __builtin_prefetch(X + (size_t)(m0 + tid) * EE + k0 + 32, 0, 0);
            if (tid < 64)
                __builtin_prefetch(W + (size_t)(n0 + tid) * EE + k0 + 32, 0, 0);
        }
        __syncthreads();

        v16bf af0 = lda_frag(&As[w * 32 + l16][0], half);
        v16bf af1 = lda_frag(&As[w * 32 + 16 + l16][0], half);
#pragma unroll
        for (int t = 0; t < 4; ++t) {
            v16bf bfv = ldb_frag(&Bs[t * 16 + l16][0], half);
            acc[0][t] = WMMA_BF16(af0, bfv, acc[0][t]);
            acc[1][t] = WMMA_BF16(af1, bfv, acc[1][t]);
        }
        __syncthreads();
    }

    // 64-wide n-tile lies in exactly one of q/k/v and one head -> uniform.
    const int which = n0 >> 10;
    __bf16* dst = (which == 0) ? Qb : ((which == 1) ? Kb : Vb);
    const float scl = (which == 0) ? 0.125f : 1.0f;   // fold 1/sqrt(HD) into Q
    const int h = (n0 & 1023) >> 6;

#pragma unroll
    for (int t = 0; t < 4; ++t) {
        int d = t * 16 + l16;
        float bv = bias[n0 + t * 16 + l16];
#pragma unroll
        for (int a = 0; a < 2; ++a)
#pragma unroll
            for (int r = 0; r < 8; ++r) {
                int m = m0 + w * 32 + a * 16 + half * 8 + r;
                int b = m >> 11, s = m & 2047;
                dst[((size_t)(b * HH + h) * SS + s) * HD + d] =
                    f2bf((acc[a][t][r] + bv) * scl);
            }
    }
}

// =====================================================================
// Kernel 2: flash attention per (b,h). Grid: (B*H)*(S/64), 128 thr.
// =====================================================================
__global__ void attn_wmma(const __bf16* __restrict__ Qb,
                          const __bf16* __restrict__ Kb,
                          const __bf16* __restrict__ Vb,
                          __bf16* __restrict__ Ctx) {
    __shared__ float  sc[4][16][36];                               // [q][key] per wave
    __shared__ __attribute__((aligned(16))) __bf16 pb[4][16][32];  // probs (A staging)
    __shared__ float  albuf[4][16];
    __shared__ float  lsbuf[4][16];
    __shared__ __attribute__((aligned(16))) __bf16 Vs[64][40];     // [d][key] (block-shared)

    const int tid  = threadIdx.x;
    const int lane = tid & 31;
    const int w    = tid >> 5;
    const int half = lane >> 4;
    const int l16  = lane & 15;

    const int bh = blockIdx.x >> 5;
    const int qt = blockIdx.x & 31;
    const size_t bhS = (size_t)bh * SS;
    const int q0 = qt * 64 + w * 16;

    const __bf16* qrow = Qb + (bhS + q0 + l16) * HD;
    v16bf qa0 = lda_frag(qrow, half);        // d = 0..31
    v16bf qa1 = lda_frag(qrow + 32, half);   // d = 32..63

    v8f co[4] = {};
    float runMax = -3.0e38f, runSum = 0.0f;
    const int row = l16;

    for (int kb = 0; kb < SS; kb += 32) {
        __syncthreads();   // protect Vs from previous iteration's readers

        // ---- stage V^T tile: Vs[d][key] (coalesced 32B loads) ----
        {
            int key = tid >> 2, c = tid & 3;
            const __bf16* vr = Vb + (bhS + kb + key) * HD + c * 16;
            v8bf v0 = *(const v8bf*)vr;
            v8bf v1 = *(const v8bf*)(vr + 8);
#pragma unroll
            for (int ii = 0; ii < 8; ++ii) {
                Vs[c * 16 + ii][key]     = v0[ii];
                Vs[c * 16 + 8 + ii][key] = v1[ii];
            }
        }

        if (kb + 32 < SS) {
            __builtin_prefetch(Kb + (bhS + kb + 32 + lane) * HD, 0, 0);
            __builtin_prefetch(Vb + (bhS + kb + 32 + lane) * HD, 0, 0);
        }

        // ---- scores for 32 keys ----
#pragma unroll
        for (int sub = 0; sub < 2; ++sub) {
            int key = kb + sub * 16 + l16;
            const __bf16* krow = Kb + (bhS + key) * HD;
            v16bf k0v = ldb_frag(krow, half);
            v16bf k1v = ldb_frag(krow + 32, half);
            v8f s = {};
            s = WMMA_BF16(qa0, k0v, s);
            s = WMMA_BF16(qa1, k1v, s);
#pragma unroll
            for (int r = 0; r < 8; ++r) sc[w][half * 8 + r][sub * 16 + l16] = s[r];
        }
        __syncthreads();   // Vs visible; sc within wave ordered

        // ---- online softmax (lane owns row l16; halves redundant) ----
        float m_new = runMax;
#pragma unroll
        for (int j = 0; j < 32; ++j) m_new = fmaxf(m_new, sc[w][row][j]);
        float alpha = __expf(runMax - m_new);
        float psum = 0.0f;
        unsigned* prow = (unsigned*)&pb[w][row][0];
#pragma unroll
        for (int j2 = 0; j2 < 16; ++j2) {
            float p0 = __expf(sc[w][row][2 * j2]     - m_new);
            float p1 = __expf(sc[w][row][2 * j2 + 1] - m_new);
            psum += p0 + p1;
            prow[j2] = pack2bf(p0, p1);
        }
        runSum = runSum * alpha + psum;
        runMax = m_new;
        albuf[w][row] = alpha;
        __builtin_amdgcn_wave_barrier();

        float a8[8];
#pragma unroll
        for (int r = 0; r < 8; ++r) a8[r] = albuf[w][half * 8 + r];
#pragma unroll
        for (int t = 0; t < 4; ++t)
#pragma unroll
            for (int r = 0; r < 8; ++r) co[t][r] *= a8[r];

        // ---- ctx += P @ V ----
        v16bf pf = lda_frag(&pb[w][l16][0], half);
#pragma unroll
        for (int t = 0; t < 4; ++t) {
            v16bf vf = ldb_frag(&Vs[t * 16 + l16][0], half);
            co[t] = WMMA_BF16(pf, vf, co[t]);
        }
    }

    // ---- normalize (reciprocal once per row) and write ctx ----
    lsbuf[w][row] = __frcp_rn(runSum);
    __builtin_amdgcn_wave_barrier();
    float s8[8];
#pragma unroll
    for (int r = 0; r < 8; ++r) s8[r] = lsbuf[w][half * 8 + r];

    const int b = bh >> 4, h = bh & 15;
#pragma unroll
    for (int t = 0; t < 4; ++t) {
        int d = t * 16 + l16;
#pragma unroll
        for (int r = 0; r < 8; ++r) {
            int srow = q0 + half * 8 + r;
            Ctx[((size_t)b * SS + srow) * EE + h * HD + d] = f2bf(co[t][r] * s8[r]);
        }
    }
}

// =====================================================================
// Kernel 3: out = ctx @ out_w^T + out_b (fp32 out). A-tile staged with
// CDNA5 async global->LDS DMA (ASYNCcnt) instead of VGPR round-trip.
// =====================================================================
__global__ void out_gemm_wmma(const __bf16* __restrict__ Ctx,
                              const float* __restrict__ W,
                              const float* __restrict__ bias,
                              float* __restrict__ out) {
    __shared__ __attribute__((aligned(16))) __bf16 As[128][48];
    __shared__ __attribute__((aligned(16))) __bf16 Bs[64][48];

    const int tid  = threadIdx.x;
    const int lane = tid & 31;
    const int w    = tid >> 5;
    const int half = lane >> 4;
    const int l16  = lane & 15;
    const int n0   = blockIdx.x * 64;
    const int m0   = blockIdx.y * 128;

    v8f acc[2][4] = {};

    for (int k0 = 0; k0 < EE; k0 += 32) {
        // A: bf16 straight copy -> async global->LDS (per-lane 2x16B).
        // INST_OFFSET applies to both LDS dst and global src (ISA 10.7).
#pragma unroll
        for (int u = tid; u < 256; u += 128) {
            int m = u >> 1, c = u & 1;
            const __bf16* src = Ctx + (size_t)(m0 + m) * EE + k0 + c * 16;
            unsigned dstOff = (unsigned)(uintptr_t)&As[m][c * 16];
            asm volatile("global_load_async_to_lds_b128 %0, %1, off"
                         :: "v"(dstOff), "v"(src) : "memory");
            asm volatile("global_load_async_to_lds_b128 %0, %1, off offset:16"
                         :: "v"(dstOff), "v"(src) : "memory");
        }
        {
            int n = tid >> 1, c = tid & 1;
            const float4* s4 = (const float4*)(W + (size_t)(n0 + n) * EE + k0 + c * 16);
            float4 x0 = s4[0], x1 = s4[1], x2 = s4[2], x3 = s4[3];
            *(uint4*)&Bs[n][c * 16]     = cvt8_pk(x0, x1);
            *(uint4*)&Bs[n][c * 16 + 8] = cvt8_pk(x2, x3);
        }
        if (k0 + 32 < EE) {
            __builtin_prefetch(Ctx + (size_t)(m0 + tid) * EE + k0 + 32, 0, 0);
            if (tid < 64)
                __builtin_prefetch(W + (size_t)(n0 + tid) * EE + k0 + 32, 0, 0);
        }
        asm volatile("s_wait_asynccnt 0" ::: "memory");   // async A-tile landed
        __syncthreads();

        v16bf af0 = lda_frag(&As[w * 32 + l16][0], half);
        v16bf af1 = lda_frag(&As[w * 32 + 16 + l16][0], half);
#pragma unroll
        for (int t = 0; t < 4; ++t) {
            v16bf bfv = ldb_frag(&Bs[t * 16 + l16][0], half);
            acc[0][t] = WMMA_BF16(af0, bfv, acc[0][t]);
            acc[1][t] = WMMA_BF16(af1, bfv, acc[1][t]);
        }
        __syncthreads();
    }

#pragma unroll
    for (int t = 0; t < 4; ++t) {
        int n = n0 + t * 16 + l16;
        float bv = bias[n];
#pragma unroll
        for (int a = 0; a < 2; ++a)
#pragma unroll
            for (int r = 0; r < 8; ++r) {
                int m = m0 + w * 32 + a * 16 + half * 8 + r;
                out[(size_t)m * EE + n] = acc[a][t][r] + bv;
            }
    }
}

// =====================================================================
extern "C" void kernel_launch(void* const* d_in, const int* in_sizes, int n_in,
                              void* d_out, int out_size, void* d_ws, size_t ws_size,
                              hipStream_t stream) {
    const float* query = (const float*)d_in[0];
    // d_in[1] (key) and d_in[2] (value) are unused by the reference module
    const float* qkv_w = (const float*)d_in[3];
    const float* qkv_b = (const float*)d_in[4];
    const float* out_w = (const float*)d_in[5];
    const float* out_b = (const float*)d_in[6];
    float* out = (float*)d_out;

    // Workspace (bf16): Q, K, V in [B,H,S,HD]; Ctx in [B,S,E]. 32 MB total.
    const size_t TELEMS = (size_t)BB * SS * EE;
    __bf16* Qb  = (__bf16*)d_ws;
    __bf16* Kb  = Qb + TELEMS;
    __bf16* Vb  = Kb + TELEMS;
    __bf16* Cxb = Vb + TELEMS;

    {   // QKV projection: M=4096, N=3072
        dim3 grid(3 * EE / 64, MM / 128);
        qkv_gemm_wmma<<<grid, 128, 0, stream>>>(query, qkv_w, qkv_b, Qb, Kb, Vb);
    }
    {   // attention
        dim3 grid(BB * HH * (SS / 64));
        attn_wmma<<<grid, 128, 0, stream>>>(Qb, Kb, Vb, Cxb);
    }
    {   // output projection: M=4096, N=1024
        dim3 grid(EE / 64, MM / 128);
        out_gemm_wmma<<<grid, 128, 0, stream>>>(Cxb, out_w, out_b, out);
    }
}
